// NEP_7249904796075
// MI455X (gfx1250) — matrix operand order repaired and checked by reading, not compiled, not run on hardware
//
#include <hip/hip_runtime.h>
#include <math.h>

// ---------------- CDNA5 (gfx1250) NEP potential fwd+bwd ----------------
// wave32; v_wmma_f32_16x16x32_f16 for the expert-MLP GEMMs (fwd + VJP).
// G tiles staged LDS-side via the Tensor Data Mover (tensor_load_to_lds +
// s_wait_tensorcnt). B operands pre-packed fragment-major (2x
// global_load_b128 per fragment), A operands built with 2x ds_load_b128.
// ds_add_f32 LDS atomics for cross-wave accumulation, global_atomic_add_f32
// for force/energy/virial scatter. Everything L2-resident on MI455X.

typedef __attribute__((ext_vector_type(16))) _Float16 v16h;
typedef __attribute__((ext_vector_type(8)))  float    v8f;
typedef __attribute__((ext_vector_type(4)))  unsigned v4u;
typedef __attribute__((ext_vector_type(8)))  int      v8i_t;
typedef __attribute__((ext_vector_type(4)))  int      v4i_t;

struct __align__(16) U4 { unsigned x, y, z, w; };
union Frag { v16h v; U4 q[2]; };

#define RCUT_F   8.0f
#define D_IN     70
#define KPAD     96      // 70 padded to 3 k-steps of 32
#define FPAD     80      // 70 padded to 5 n-tiles of 16
#define ASTRIDE  72      // row stride of dE/dg coef array
#define N_ELEM   4
#define HIDDEN   128
#define MAXNN    32
#define NT1      (HIDDEN / 16)   // 8
#define KT1      (KPAD / 32)     // 3
#define FT2      (FPAD / 16)     // 5
#define KT2      (HIDDEN / 32)   // 4
#define PI_F     3.14159265358979323846f

__device__ __forceinline__ void gAtomicAdd(float* p, float v) {
    unsafeAtomicAdd(p, v);   // -> global_atomic_add_f32
}

__device__ __forceinline__ unsigned pack2h(float lo, float hi) {
    union { _Float16 h[2]; unsigned u; } c;
    c.h[0] = (_Float16)lo; c.h[1] = (_Float16)hi;
    return c.u;
}

__device__ __forceinline__ float fast_tanh(float x) {
    float t = __expf(2.0f * x);                       // v_exp_f32
    return 1.0f - 2.0f * __builtin_amdgcn_rcpf(t + 1.0f);  // v_rcp_f32, no div expansion
}

// ---------------- K0: zero the accumulated output regions ----------------
__global__ void k_zero(float* __restrict__ p, int n) {
    int i = blockIdx.x * blockDim.x + threadIdx.x;
    if (i < n) p[i] = 0.0f;
}

// ---------------- K_prep: pack W0 into fragment-major f16 B operands -----
// B1p[e][nt][kt][lane][v] : pair (k,k+1) of B1[k][n] = W0[e][n][k]
// B2p[e][ft][kt][lane][v] : pair (k,k+1) of B2[k][n] = W0[e][k][n]
__global__ void k_prep(const float* __restrict__ W0,
                       unsigned* __restrict__ B1p, unsigned* __restrict__ B2p) {
    const int nf1 = N_ELEM * NT1 * KT1 * 32 * 8;   // 24576
    const int nf2 = N_ELEM * FT2 * KT2 * 32 * 8;   // 20480
    int idx = blockIdx.x * blockDim.x + threadIdx.x;
    if (idx < nf1) {
        int t = idx;
        int v    = t & 7;  t >>= 3;
        int lane = t & 31; t >>= 5;
        int kt   = t % 3;  t /= 3;
        int nt   = t & 7;  t >>= 3;
        int e    = t;
        int k = kt * 32 + (v >> 2) * 16 + (lane >> 4) * 8 + 2 * (v & 3);
        int n = nt * 16 + (lane & 15);
        float lo = (k     < D_IN) ? W0[(e * HIDDEN + n) * D_IN + k]     : 0.0f;
        float hi = (k + 1 < D_IN) ? W0[(e * HIDDEN + n) * D_IN + k + 1] : 0.0f;
        B1p[idx] = pack2h(lo, hi);
    } else if (idx - nf1 < nf2) {
        int t = idx - nf1;
        int i2 = t;
        int v    = t & 7;  t >>= 3;
        int lane = t & 31; t >>= 5;
        int kt   = t & 3;  t >>= 2;
        int ft   = t % 5;  t /= 5;
        int e    = t;
        int k = kt * 32 + (v >> 2) * 16 + (lane >> 4) * 8 + 2 * (v & 3);   // hidden index, always < 128
        int n = ft * 16 + (lane & 15);                // feature index
        float lo = (n < D_IN) ? W0[(e * HIDDEN + k)     * D_IN + n] : 0.0f;
        float hi = (n < D_IN) ? W0[(e * HIDDEN + k + 1) * D_IN + n] : 0.0f;
        B2p[i2] = pack2h(lo, hi);
    }
}

// ---------------- K1: radial Chebyshev descriptor -> Gh (f16, scaled) ----
__global__ void __launch_bounds__(128)
k_desc(const float* __restrict__ pos, const float* __restrict__ qscal,
       const int* __restrict__ nbr, int N, unsigned* __restrict__ Ghu) {
    int i = blockIdx.x * blockDim.x + threadIdx.x;
    if (i >= N) return;
    float px = pos[3 * i], py = pos[3 * i + 1], pz = pos[3 * i + 2];
    float f[D_IN];
#pragma unroll
    for (int k = 0; k < D_IN; ++k) f[k] = 0.0f;
    for (int nn = 0; nn < MAXNN; ++nn) {
        int j = nbr[i * MAXNN + nn];
        float dx = pos[3 * j] - px, dy = pos[3 * j + 1] - py, dz = pos[3 * j + 2] - pz;
        float d = sqrtf(dx * dx + dy * dy + dz * dz + 1e-12f);
        bool valid = (d < RCUT_F) && (j != i);
        if (!valid) continue;
        float fc = 0.5f * (__cosf(PI_F * d * (1.0f / RCUT_F)) + 1.0f);
        float x = 2.0f * d * (1.0f / RCUT_F) - 1.0f;
        x = fminf(1.0f, fmaxf(-1.0f, x));
        float Tp = 1.0f, Tc = x;
        f[0] += fc;
        f[1] += fc * x;
#pragma unroll
        for (int k = 2; k < D_IN; ++k) {
            float Tn = 2.0f * x * Tc - Tp;
            f[k] += fc * Tn;
            Tp = Tc; Tc = Tn;
        }
    }
    unsigned* row = Ghu + (size_t)i * (KPAD / 2);
#pragma unroll
    for (int k = 0; k < D_IN / 2; ++k)
        row[k] = pack2h(f[2 * k] * qscal[2 * k], f[2 * k + 1] * qscal[2 * k + 1]);
#pragma unroll
    for (int k = D_IN / 2; k < KPAD / 2; ++k) row[k] = 0u;
}

// ---------------- K2: WMMA expert MLP, forward + VJP wrt g ---------------
// One workgroup = 128 threads = 4 waves = 4 experts, 16-atom tile.
__global__ void __launch_bounds__(128)
k_mlp(const U4* __restrict__ Gq, const U4* __restrict__ B1p,
      const U4* __restrict__ B2p, const float* __restrict__ b0,
      const float* __restrict__ W1, const float* __restrict__ qscal,
      const float* __restrict__ sbias, const int* __restrict__ types,
      const int* __restrict__ sids, int N,
      float* __restrict__ Acoef, float* __restrict__ e_atom,
      float* __restrict__ e_total) {
    __shared__ __align__(16) _Float16 sG[16][KPAD];
    __shared__ __align__(16) _Float16 sU[16][HIDDEN];
    __shared__ float sDG[16][FPAD];
    __shared__ float sOut[16];
    __shared__ int   sType[16];

    const int tid   = threadIdx.x;
    const int lane  = tid & 31;
    const int wave  = tid >> 5;           // expert index e
    const int base  = blockIdx.x * 16;
    const int mrow  = lane & 15;
    const int khalf = lane >> 4;
    const U4  zq = {0u, 0u, 0u, 0u};

    // ---- TDM: DMA the contiguous 3KB G tile (16 rows x 192B) into LDS ----
    if (wave == 0) {   // uniform branch: one TDM issue per workgroup
        unsigned ldsoff = (unsigned)(size_t)(void*)&sG[0][0];
        unsigned long long ga =
            (unsigned long long)(size_t)(const void*)Gq + (unsigned long long)base * (KPAD * 2);
        v4u g0;
        g0[0] = 1u;                                         // count = 1 (valid D#)
        g0[1] = ldsoff;                                     // lds_addr
        g0[2] = (unsigned)(ga & 0xffffffffu);               // global_addr[31:0]
        g0[3] = (unsigned)((ga >> 32) & 0x01ffffffu)        // global_addr[56:32]
                | 0x80000000u;                              // type = 2 ("image")
        v8i_t g1;
        g1[0] = 0x00030000;          // data_size = 3 (8B), no multicast/pad/iterate
        g1[1] = (int)(384u << 16);   // tensor_dim0 = 384 (x8B = 3072B)
        g1[2] = 0x00010000;          // tensor_dim1 = 1
        g1[3] = (int)(384u << 16);   // tile_dim0 = 384
        g1[4] = 0;                   // tile_dim1/2 unused
        g1[5] = 384;                 // tensor_dim0_stride = 384
        g1[6] = 0;
        g1[7] = 0;
        v4i_t gz4 = {0, 0, 0, 0};
#if __clang_major__ >= 23
        v8i_t gz8 = {0, 0, 0, 0, 0, 0, 0, 0};
        __builtin_amdgcn_tensor_load_to_lds(g0, g1, gz4, gz4, gz8, 0);
#else
        __builtin_amdgcn_tensor_load_to_lds(g0, g1, gz4, gz4, 0);
#endif
        __builtin_amdgcn_s_wait_tensorcnt(0);               // s_wait_tensorcnt 0
    }
    for (int idx = tid; idx < 16 * FPAD; idx += 128) sDG[idx / FPAD][idx % FPAD] = 0.0f;
    if (tid < 16) {
        int gm = base + tid;
        sOut[tid]  = 0.0f;
        sType[tid] = (gm < N) ? types[gm] : -1;
    }
    __syncthreads();

    const int e = wave;
    // ---- GEMM1: pre[16,128] = Gtile[16,96] x B1[e][96,128] ----
    for (int nt = 0; nt < NT1; ++nt) {
        v8f acc = {0.f, 0.f, 0.f, 0.f, 0.f, 0.f, 0.f, 0.f};
        for (int kt = 0; kt < KT1; ++kt) {
            Frag fa, fb;
            // A: 2x ds_load_b128 (contiguous 8-half runs)
            fa.q[0] = *(const U4*)&sG[mrow][kt * 32 + khalf * 8];
            fa.q[1] = *(const U4*)&sG[mrow][kt * 32 + 16 + khalf * 8];
            // B: 2x global_load_b128, fragment-major packed
            const U4* bp = B1p + (size_t)((((e * NT1 + nt) * KT1 + kt) * 32 + lane) << 1);
            fb.q[0] = bp[0];
            fb.q[1] = bp[1];
            acc = __builtin_amdgcn_wmma_f32_16x16x32_f16(false, fa.v, false, fb.v,
                                                         (short)0, acc, false, false);
        }
        int   ng   = nt * 16 + mrow;
        float bias = b0[e * HIDDEN + ng];
        float w1   = W1[e * HIDDEN + ng];
#pragma unroll
        for (int r = 0; r < 8; ++r) {
            int   m = r + 8 * khalf;
            float h = fast_tanh(acc[r] + bias);
            if (sType[m] == e) {
                atomicAdd(&sOut[m], w1 * h);                   // ds_add_f32
                sU[m][ng] = (_Float16)(w1 * (1.0f - h * h));
            }
        }
    }
    __syncthreads();

    if (tid < 16) {
        int gm = base + tid;
        if (gm < N) {
            float ea = sOut[tid] - sbias[0];
            e_atom[gm] = ea;
            gAtomicAdd(&e_total[sids[gm]], ea);
        }
    }

    // ---- GEMM2: dG[16,80] = sum_e mask_e(U)[16,128] x B2[e][128,80] ----
    const bool rowon = (sType[mrow] == e);
    for (int ft = 0; ft < FT2; ++ft) {
        v8f acc = {0.f, 0.f, 0.f, 0.f, 0.f, 0.f, 0.f, 0.f};
        for (int kt = 0; kt < KT2; ++kt) {
            Frag fa, fb;
            U4 a0 = *(const U4*)&sU[mrow][kt * 32 + khalf * 8];
            U4 a1 = *(const U4*)&sU[mrow][kt * 32 + 16 + khalf * 8];
            fa.q[0] = rowon ? a0 : zq;
            fa.q[1] = rowon ? a1 : zq;
            const U4* bp = B2p + (size_t)((((e * FT2 + ft) * KT2 + kt) * 32 + lane) << 1);
            fb.q[0] = bp[0];
            fb.q[1] = bp[1];
            acc = __builtin_amdgcn_wmma_f32_16x16x32_f16(false, fa.v, false, fb.v,
                                                         (short)0, acc, false, false);
        }
#pragma unroll
        for (int r = 0; r < 8; ++r) {
            int m = r + 8 * khalf;
            atomicAdd(&sDG[m][ft * 16 + mrow], acc[r]);        // ds_add_f32
        }
    }
    __syncthreads();

    // chain rule through q_scaler; write dE/d(raw feature)
    for (int idx = tid; idx < 16 * D_IN; idx += 128) {
        int m = idx / D_IN, k = idx % D_IN;
        int gm = base + m;
        if (gm < N) Acoef[(size_t)gm * ASTRIDE + k] = sDG[m][k] * qscal[k];
    }
}

// ---------------- K3: descriptor backward + force scatter ----------------
// 256 threads = 8 atoms x 32 neighbor lanes; wave w owns atom w (broadcast LDS reads).
__global__ void __launch_bounds__(256)
k_force(const float* __restrict__ pos, const float* __restrict__ Acoef,
        const int* __restrict__ nbr, int N, float* __restrict__ F) {
    __shared__ float sA[8][D_IN];
    __shared__ float sP[8][3];
    const int tid  = threadIdx.x;
    const int il   = tid >> 5;
    const int lane = tid & 31;
    const int base = blockIdx.x * 8;

    for (int idx = tid; idx < 8 * D_IN; idx += 256) {
        int m = idx / D_IN, k = idx % D_IN;
        int gm = base + m;
        sA[m][k] = (gm < N) ? Acoef[(size_t)gm * ASTRIDE + k] : 0.0f;
    }
    if (tid < 24) {
        int m = tid / 3, c = tid % 3;
        int gm = base + m;
        sP[m][c] = (gm < N) ? pos[gm * 3 + c] : 0.0f;
    }
    __syncthreads();

    int i = base + il;
    float vx = 0.0f, vy = 0.0f, vz = 0.0f;
    if (i < N) {
        int   j  = nbr[i * MAXNN + lane];
        float dx = pos[3 * j]     - sP[il][0];
        float dy = pos[3 * j + 1] - sP[il][1];
        float dz = pos[3 * j + 2] - sP[il][2];
        float d  = sqrtf(dx * dx + dy * dy + dz * dz + 1e-12f);
        bool valid = (d < RCUT_F) && (j != i);
        if (valid) {
            float inv  = 1.0f / d;
            float ang  = PI_F * d * (1.0f / RCUT_F);
            float fc   = 0.5f * (__cosf(ang) + 1.0f);
            float dfc  = -0.5f * (PI_F / RCUT_F) * __sinf(ang);
            float x    = 2.0f * d * (1.0f / RCUT_F) - 1.0f;
            float dxdd = 0.25f;
            if (x > 1.0f)       { x = 1.0f;  dxdd = 0.0f; }
            else if (x < -1.0f) { x = -1.0f; dxdd = 0.0f; }
            float Tp = 1.0f, Tc = x, Tpd = 0.0f, Tcd = 1.0f;
            float acc = sA[il][0] * dfc + sA[il][1] * (dfc * x + fc * dxdd);
#pragma unroll
            for (int k = 2; k < D_IN; ++k) {
                float Tn  = 2.0f * x * Tc - Tp;
                float Tnd = 2.0f * Tc + 2.0f * x * Tcd - Tpd;
                acc += sA[il][k] * (dfc * Tn + fc * Tnd * dxdd);
                Tp = Tc; Tc = Tn; Tpd = Tcd; Tcd = Tnd;
            }
            float s = acc * inv;       // dE/dd / d
            vx = s * dx; vy = s * dy; vz = s * dz;
            // grad_rj = +v  ->  forces[j] -= v
            gAtomicAdd(&F[3 * j],     -vx);
            gAtomicAdd(&F[3 * j + 1], -vy);
            gAtomicAdd(&F[3 * j + 2], -vz);
        }
    }
    // grad_ri = -v  ->  forces[i] += v  (wave-reduce 32 neighbor lanes)
    for (int off = 16; off > 0; off >>= 1) {
        vx += __shfl_xor(vx, off, 32);
        vy += __shfl_xor(vy, off, 32);
        vz += __shfl_xor(vz, off, 32);
    }
    if (lane == 0 && i < N) {
        gAtomicAdd(&F[3 * i],     vx);
        gAtomicAdd(&F[3 * i + 1], vy);
        gAtomicAdd(&F[3 * i + 2], vz);
    }
}

// ---------------- K4: virial = -segment_sum(p (x) f) ---------------------
__global__ void __launch_bounds__(256)
k_virial(const float* __restrict__ pos, const float* __restrict__ F,
         const int* __restrict__ sids, int N, float* __restrict__ vir) {
    int i = blockIdx.x * blockDim.x + threadIdx.x;
    if (i >= N) return;
    int s = sids[i];
    float p[3] = {pos[3 * i], pos[3 * i + 1], pos[3 * i + 2]};
    float f[3] = {F[3 * i], F[3 * i + 1], F[3 * i + 2]};
#pragma unroll
    for (int a = 0; a < 3; ++a)
#pragma unroll
        for (int b = 0; b < 3; ++b)
            gAtomicAdd(&vir[s * 9 + a * 3 + b], -p[a] * f[b]);
}

// -------------------------------------------------------------------------
extern "C" void kernel_launch(void* const* d_in, const int* in_sizes, int n_in,
                              void* d_out, int out_size, void* d_ws, size_t ws_size,
                              hipStream_t stream) {
    const float* pos   = (const float*)d_in[0];
    const float* qscal = (const float*)d_in[1];
    const float* W0    = (const float*)d_in[2];
    const float* b0    = (const float*)d_in[3];
    const float* W1    = (const float*)d_in[4];
    const float* sbias = (const float*)d_in[5];
    const int*   types = (const int*)d_in[6];
    const int*   nbr   = (const int*)d_in[7];
    const int*   sids  = (const int*)d_in[8];

    const int N = in_sizes[6];
    const int S = (out_size - 4 * N) / 10;   // e_atom N + e_total S + forces 3N + virial 9S

    float* out     = (float*)d_out;
    float* e_atom  = out;
    float* e_total = out + N;
    float* forces  = out + N + S;
    float* virial  = out + N + S + 3 * (size_t)N;

    char*  ws  = (char*)d_ws;
    size_t off = 0;
    unsigned* Ghu = (unsigned*)(ws + off);
    off += (size_t)N * KPAD * sizeof(_Float16);               off = (off + 255) & ~(size_t)255;
    float* Acoef = (float*)(ws + off);
    off += (size_t)N * ASTRIDE * sizeof(float);               off = (off + 255) & ~(size_t)255;
    unsigned* B1p = (unsigned*)(ws + off);
    off += (size_t)(N_ELEM * NT1 * KT1 * 32 * 8) * 4;         off = (off + 255) & ~(size_t)255;
    unsigned* B2p = (unsigned*)(ws + off);

    const int zn = S + 3 * N + 9 * S;
    k_zero<<<(zn + 255) / 256, 256, 0, stream>>>(e_total, zn);

    const int prep_total = N_ELEM * NT1 * KT1 * 32 * 8 + N_ELEM * FT2 * KT2 * 32 * 8;
    k_prep<<<(prep_total + 255) / 256, 256, 0, stream>>>(W0, B1p, B2p);

    k_desc<<<(N + 127) / 128, 128, 0, stream>>>(pos, qscal, nbr, N, Ghu);

    k_mlp<<<(N + 15) / 16, 128, 0, stream>>>((const U4*)Ghu, (const U4*)B1p, (const U4*)B2p,
                                             b0, W1, qscal, sbias, types, sids, N,
                                             Acoef, e_atom, e_total);

    k_force<<<(N + 7) / 8, 256, 0, stream>>>(pos, Acoef, nbr, N, forces);

    k_virial<<<(N + 255) / 256, 256, 0, stream>>>(pos, forces, sids, N, virial);
}
